// LSTMSoftmax_57483842290153
// MI455X (gfx1250) — matrix operand order repaired and checked by reading, not compiled, not run on hardware
//
#include <hip/hip_runtime.h>
#include <hip/hip_bf16.h>
#include <math.h>

#define H_DIM  1024
#define IN_DIM 512
#define OUT_DIM 512
#define B_DIM  4096
#define G4H    4096   // 4*H

typedef __attribute__((ext_vector_type(16))) __bf16 v16bf;
typedef __attribute__((ext_vector_type(8)))  float  v8f;

__device__ __forceinline__ float sigmoidf_(float x) { return 1.0f / (1.0f + __expf(-x)); }

__device__ __forceinline__ float bflo(uint32_t u) { return __uint_as_float(u << 16); }
__device__ __forceinline__ float bfhi(uint32_t u) { return __uint_as_float(u & 0xffff0000u); }

__device__ __forceinline__ float wave_allsum(float v) {
#pragma unroll
    for (int off = 16; off >= 1; off >>= 1) v += __shfl_xor(v, off, 32);
    return v;
}

// -------- device-wide sense barrier (persistent kernel) --------
__device__ __forceinline__ void grid_barrier(int* bar, int nblk) {
    __syncthreads();
    if (threadIdx.x == 0) {
        int* cnt = bar;
        int* gen = bar + 1;
        __threadfence();  // make our global writes visible before arriving
        int g = __hip_atomic_load(gen, __ATOMIC_RELAXED, __HIP_MEMORY_SCOPE_AGENT);
        if (__hip_atomic_fetch_add(cnt, 1, __ATOMIC_ACQ_REL, __HIP_MEMORY_SCOPE_AGENT) == nblk - 1) {
            __hip_atomic_store(cnt, 0, __ATOMIC_RELAXED, __HIP_MEMORY_SCOPE_AGENT);
            __hip_atomic_fetch_add(gen, 1, __ATOMIC_RELEASE, __HIP_MEMORY_SCOPE_AGENT);
        } else {
            while (__hip_atomic_load(gen, __ATOMIC_ACQUIRE, __HIP_MEMORY_SCOPE_AGENT) == g)
                __builtin_amdgcn_s_sleep(2);
        }
    }
    __syncthreads();
    // per-wave acquire so every CU's cache sees cross-WGP h writes
    (void)__hip_atomic_load(bar + 1, __ATOMIC_ACQUIRE, __HIP_MEMORY_SCOPE_AGENT);
}

__global__ void init_bar_kernel(int* bar) {
    if (threadIdx.x < 2) bar[threadIdx.x] = 0;
}

// -------- fp32 -> packed bf16 (2 per dword) conversion --------
__global__ void cvt2bf16_kernel(const float* __restrict__ in, uint32_t* __restrict__ out, int n2) {
    int i = blockIdx.x * blockDim.x + threadIdx.x;
    if (i < n2) {
        float2 v = ((const float2*)in)[i];
        unsigned short lo = __builtin_bit_cast(unsigned short, (__bf16)v.x);
        unsigned short hi = __builtin_bit_cast(unsigned short, (__bf16)v.y);
        out[i] = ((uint32_t)hi << 16) | lo;
    }
}

// -------- WMMA GEMM: D[m,n] = sum_k A[m,k]*W[n,k] + bias0[n] (+bias1[n]) --------
// One wave per 16x16 tile of D; K consumed 32 at a time via v_wmma_f32_16x16x32_bf16.
__global__ void wmma_gemm_bias(const float* __restrict__ A, const float* __restrict__ W,
                               const float* __restrict__ bias0, const float* __restrict__ bias1,
                               float* __restrict__ D, int M, int N, int K) {
    const int wave  = threadIdx.x >> 5;
    const int lane  = threadIdx.x & 31;
    const int tile  = blockIdx.x * (blockDim.x >> 5) + wave;
    const int tilesN = N >> 4;
    const int mt = (tile / tilesN) << 4;
    const int nt = (tile % tilesN) << 4;
    const int l16  = lane & 15;
    const int half = lane >> 4;

    const float* arow = A + (size_t)(mt + l16) * K;   // lane's A row (M = l16)
    const float* wrow = W + (size_t)(nt + l16) * K;   // lane's B column (N = l16), row-major W[N,K]
    float bias = bias0[nt + l16] + (bias1 ? bias1[nt + l16] : 0.0f);

    v8f acc;
#pragma unroll
    for (int i = 0; i < 8; ++i) acc[i] = bias;

#pragma unroll 2
    for (int k0 = 0; k0 < K; k0 += 32) {
        v16bf a, b;
        // A 16x32 bf16: lanes 0-15 hold K=0..15, lanes 16-31 hold K=16..31
#pragma unroll
        for (int i = 0; i < 16; ++i) a[i] = (__bf16)arow[k0 + half * 16 + i];
        // B 32x16 bf16: lanes 0-15 hold K=0..15, lanes 16-31 hold K=16..31
#pragma unroll
        for (int i = 0; i < 16; ++i) b[i] = (__bf16)wrow[k0 + half * 16 + i];
        acc = __builtin_amdgcn_wmma_f32_16x16x32_bf16(false, a, false, b, (short)0, acc,
                                                      false, false);
    }
    // D 16x16 f32: VGPR v -> row (half*8 + v), lane -> col l16
#pragma unroll
    for (int v = 0; v < 8; ++v)
        D[(size_t)(mt + half * 8 + v) * N + (nt + l16)] = acc[v];
}

// -------- persistent recurrent kernel: one wave per hidden unit j --------
// Weights pre-packed bf16 (2/dword, 512 dwords per row); h vectors stay fp32.
__global__ void lstm_recurrent(const float* __restrict__ G0,      // [B,4H] x-proj + biases (f32)
                               const uint32_t* __restrict__ Wb0,  // Whh0 packed bf16 [4H][512]
                               const uint32_t* __restrict__ Wbi1, // Wih1 packed bf16
                               const uint32_t* __restrict__ Wbh1, // Whh1 packed bf16
                               const float* __restrict__ bih1, const float* __restrict__ bhh1,
                               const float* __restrict__ h_init, // [2,H]
                               const float* __restrict__ c_init, // [2,H]
                               float* __restrict__ h0buf,        // [2,H] double buffer
                               float* __restrict__ h1buf,        // [2,H] double buffer
                               float* __restrict__ H1out,        // [B,H]
                               int* __restrict__ bar) {
    const int lane = threadIdx.x & 31;
    const int j    = (blockIdx.x * blockDim.x + threadIdx.x) >> 5;  // hidden unit, 0..1023
    const int nblk = gridDim.x;
    const int KW   = H_DIM / 2;  // 512 packed dwords per row

    const uint32_t* w00 = Wb0 + (size_t)j * KW;
    const uint32_t* w01 = Wb0 + (size_t)(H_DIM + j) * KW;
    const uint32_t* w02 = Wb0 + (size_t)(2 * H_DIM + j) * KW;
    const uint32_t* w03 = Wb0 + (size_t)(3 * H_DIM + j) * KW;
    const uint32_t* wi0 = Wbi1 + (size_t)j * KW;
    const uint32_t* wi1 = Wbi1 + (size_t)(H_DIM + j) * KW;
    const uint32_t* wi2 = Wbi1 + (size_t)(2 * H_DIM + j) * KW;
    const uint32_t* wi3 = Wbi1 + (size_t)(3 * H_DIM + j) * KW;
    const uint32_t* wh0 = Wbh1 + (size_t)j * KW;
    const uint32_t* wh1 = Wbh1 + (size_t)(H_DIM + j) * KW;
    const uint32_t* wh2 = Wbh1 + (size_t)(2 * H_DIM + j) * KW;
    const uint32_t* wh3 = Wbh1 + (size_t)(3 * H_DIM + j) * KW;

    const float bi_i = bih1[j] + bhh1[j];
    const float bi_f = bih1[H_DIM + j] + bhh1[H_DIM + j];
    const float bi_g = bih1[2 * H_DIM + j] + bhh1[2 * H_DIM + j];
    const float bi_o = bih1[3 * H_DIM + j] + bhh1[3 * H_DIM + j];

    float c0 = c_init[j];           // identical in all lanes; stays coherent
    float c1 = c_init[H_DIM + j];

    if (lane == 0) {
        h0buf[j] = h_init[j];
        h1buf[j] = h_init[H_DIM + j];
    }
    grid_barrier(bar, nblk);

    for (int t = 0; t < B_DIM; ++t) {
        const int p = t & 1;
        // ---- layer 0: gates = G0[t] + Whh0 @ h0_prev ----
        const float2* h0r = (const float2*)(h0buf + p * H_DIM);
        float a0 = 0.f, a1 = 0.f, a2 = 0.f, a3 = 0.f;
#pragma unroll 4
        for (int k2 = lane; k2 < KW; k2 += 32) {
            float2 hv = h0r[k2];
            uint32_t u0 = __builtin_nontemporal_load(w00 + k2);
            uint32_t u1 = __builtin_nontemporal_load(w01 + k2);
            uint32_t u2 = __builtin_nontemporal_load(w02 + k2);
            uint32_t u3 = __builtin_nontemporal_load(w03 + k2);
            a0 = fmaf(bflo(u0), hv.x, fmaf(bfhi(u0), hv.y, a0));
            a1 = fmaf(bflo(u1), hv.x, fmaf(bfhi(u1), hv.y, a1));
            a2 = fmaf(bflo(u2), hv.x, fmaf(bfhi(u2), hv.y, a2));
            a3 = fmaf(bflo(u3), hv.x, fmaf(bfhi(u3), hv.y, a3));
        }
        a0 = wave_allsum(a0); a1 = wave_allsum(a1);
        a2 = wave_allsum(a2); a3 = wave_allsum(a3);

        const float* g = G0 + (size_t)t * G4H;
        float iv = sigmoidf_(g[j] + a0);
        float fv = sigmoidf_(g[H_DIM + j] + a1);
        float gv = tanhf(g[2 * H_DIM + j] + a2);
        float ov = sigmoidf_(g[3 * H_DIM + j] + a3);
        c0 = fv * c0 + iv * gv;
        float h0n = ov * tanhf(c0);
        if (lane == 0) h0buf[(1 - p) * H_DIM + j] = h0n;

        grid_barrier(bar, nblk);  // h0n fully published; one barrier per step suffices

        // ---- layer 1: gates = Wih1 @ h0n + Whh1 @ h1_prev + b ----
        const float2* xr  = (const float2*)(h0buf + (1 - p) * H_DIM);
        const float2* h1r = (const float2*)(h1buf + p * H_DIM);
        float s0 = 0.f, s1 = 0.f, s2 = 0.f, s3 = 0.f;
        float r0 = 0.f, r1 = 0.f, r2 = 0.f, r3 = 0.f;
#pragma unroll 2
        for (int k2 = lane; k2 < KW; k2 += 32) {
            float2 xv = xr[k2];
            float2 hv = h1r[k2];
            uint32_t ui0 = __builtin_nontemporal_load(wi0 + k2);
            uint32_t ui1 = __builtin_nontemporal_load(wi1 + k2);
            uint32_t ui2 = __builtin_nontemporal_load(wi2 + k2);
            uint32_t ui3 = __builtin_nontemporal_load(wi3 + k2);
            uint32_t uh0 = __builtin_nontemporal_load(wh0 + k2);
            uint32_t uh1 = __builtin_nontemporal_load(wh1 + k2);
            uint32_t uh2 = __builtin_nontemporal_load(wh2 + k2);
            uint32_t uh3 = __builtin_nontemporal_load(wh3 + k2);
            s0 = fmaf(bflo(ui0), xv.x, fmaf(bfhi(ui0), xv.y, s0));
            s1 = fmaf(bflo(ui1), xv.x, fmaf(bfhi(ui1), xv.y, s1));
            s2 = fmaf(bflo(ui2), xv.x, fmaf(bfhi(ui2), xv.y, s2));
            s3 = fmaf(bflo(ui3), xv.x, fmaf(bfhi(ui3), xv.y, s3));
            r0 = fmaf(bflo(uh0), hv.x, fmaf(bfhi(uh0), hv.y, r0));
            r1 = fmaf(bflo(uh1), hv.x, fmaf(bfhi(uh1), hv.y, r1));
            r2 = fmaf(bflo(uh2), hv.x, fmaf(bfhi(uh2), hv.y, r2));
            r3 = fmaf(bflo(uh3), hv.x, fmaf(bfhi(uh3), hv.y, r3));
        }
        s0 = wave_allsum(s0); s1 = wave_allsum(s1);
        s2 = wave_allsum(s2); s3 = wave_allsum(s3);
        r0 = wave_allsum(r0); r1 = wave_allsum(r1);
        r2 = wave_allsum(r2); r3 = wave_allsum(r3);

        float i1 = sigmoidf_(s0 + r0 + bi_i);
        float f1 = sigmoidf_(s1 + r1 + bi_f);
        float g1 = tanhf(s2 + r2 + bi_g);
        float o1 = sigmoidf_(s3 + r3 + bi_o);
        c1 = f1 * c1 + i1 * g1;
        float h1n = o1 * tanhf(c1);
        if (lane == 0) {
            h1buf[(1 - p) * H_DIM + j] = h1n;
            __builtin_nontemporal_store(h1n, &H1out[(size_t)t * H_DIM + j]);
        }
    }
}

// -------- softmax over rows of 512 --------
__global__ void softmax512(const float* __restrict__ in, float* __restrict__ out) {
    const int r = blockIdx.x;
    const float* row = in + (size_t)r * OUT_DIM;
    __shared__ float red[256];
    int t = threadIdx.x;
    float a = row[t], b = row[t + 256];
    red[t] = fmaxf(a, b);
    __syncthreads();
    for (int s = 128; s > 0; s >>= 1) {
        if (t < s) red[t] = fmaxf(red[t], red[t + s]);
        __syncthreads();
    }
    float mx = red[0];
    __syncthreads();
    float ea = __expf(a - mx), eb = __expf(b - mx);
    red[t] = ea + eb;
    __syncthreads();
    for (int s = 128; s > 0; s >>= 1) {
        if (t < s) red[t] += red[t + s];
        __syncthreads();
    }
    float inv = 1.0f / red[0];
    out[(size_t)r * OUT_DIM + t]       = ea * inv;
    out[(size_t)r * OUT_DIM + t + 256] = eb * inv;
}

extern "C" void kernel_launch(void* const* d_in, const int* in_sizes, int n_in,
                              void* d_out, int out_size, void* d_ws, size_t ws_size,
                              hipStream_t stream) {
    (void)in_sizes; (void)n_in; (void)out_size; (void)ws_size;
    const float* x     = (const float*)d_in[0];   // [B, IN]
    const float* Wih0  = (const float*)d_in[1];   // [4H, IN]
    const float* Whh0  = (const float*)d_in[2];   // [4H, H]
    const float* bih0  = (const float*)d_in[3];
    const float* bhh0  = (const float*)d_in[4];
    const float* Wih1  = (const float*)d_in[5];   // [4H, H]
    const float* Whh1  = (const float*)d_in[6];   // [4H, H]
    const float* bih1  = (const float*)d_in[7];
    const float* bhh1  = (const float*)d_in[8];
    const float* h0    = (const float*)d_in[9];   // [2, H]
    const float* c0    = (const float*)d_in[10];  // [2, H]
    const float* fc_w  = (const float*)d_in[11];  // [OUT, H]
    const float* fc_b  = (const float*)d_in[12];
    float* out = (float*)d_out;

    // workspace layout
    float*    G0    = (float*)d_ws;                        // [B,4H]   64 MB
    float*    H1    = G0 + (size_t)B_DIM * G4H;            // [B,H]    16 MB
    float*    LG    = H1 + (size_t)B_DIM * H_DIM;          // [B,OUT]   8 MB
    float*    h0buf = LG + (size_t)B_DIM * OUT_DIM;        // [2,H]
    float*    h1buf = h0buf + 2 * H_DIM;                   // [2,H]
    int*      bar   = (int*)(h1buf + 2 * H_DIM);           // 2 ints (pad to 4)
    uint32_t* Wb0   = (uint32_t*)(bar + 4);                // Whh0 bf16 [4H][512]  8 MB
    uint32_t* Wbi1  = Wb0 + (size_t)G4H * (H_DIM / 2);     // Wih1 bf16            8 MB
    uint32_t* Wbh1  = Wbi1 + (size_t)G4H * (H_DIM / 2);    // Whh1 bf16            8 MB

    init_bar_kernel<<<1, 32, 0, stream>>>(bar);

    // 0) pack the three recurrent weight matrices to bf16 (2/dword)
    {
        const int n2 = G4H * H_DIM / 2;  // 2M dwords each
        cvt2bf16_kernel<<<(n2 + 255) / 256, 256, 0, stream>>>(Whh0, Wb0, n2);
        cvt2bf16_kernel<<<(n2 + 255) / 256, 256, 0, stream>>>(Wih1, Wbi1, n2);
        cvt2bf16_kernel<<<(n2 + 255) / 256, 256, 0, stream>>>(Whh1, Wbh1, n2);
    }

    // 1) G0 = x @ Wih0^T + (bih0 + bhh0)   [4096 x 4096], K=512  (WMMA)
    {
        int tiles = (B_DIM / 16) * (G4H / 16);  // 65536
        wmma_gemm_bias<<<tiles / 8, 256, 0, stream>>>(x, Wih0, bih0, bhh0, G0,
                                                      B_DIM, G4H, IN_DIM);
    }

    // 2) persistent recurrent sweep: 128 blocks x 256 thr = 1024 waves = H
    lstm_recurrent<<<128, 256, 0, stream>>>(G0, Wb0, Wbi1, Wbh1, bih1, bhh1,
                                            h0, c0, h0buf, h1buf, H1, bar);

    // 3) logits = H1 @ fc_w^T + fc_b   [4096 x 512], K=1024  (WMMA)
    {
        int tiles = (B_DIM / 16) * (OUT_DIM / 16);  // 8192
        wmma_gemm_bias<<<tiles / 8, 256, 0, stream>>>(H1, fc_w, fc_b, nullptr, LG,
                                                      B_DIM, OUT_DIM, H_DIM);
    }

    // 4) softmax per row
    softmax512<<<B_DIM, 256, 0, stream>>>(LG, out);
}
